// SimpleRNN_2181843386833
// MI455X (gfx1250) — matrix operand order repaired and checked by reading, not compiled, not run on hardware
//
#include <hip/hip_runtime.h>
#include <hip/hip_bf16.h>

// ---------------------------------------------------------------------------
// Seq2seq bi-LSTM for MI455X (gfx1250, wave32, WMMA).
//   B=128, S=256, D=256, H=1024, T=128, 4H=4096.
// bf16 operands + f32 accumulation via v_wmma_f32_16x16x32_bf16.
// Per recurrent step: fused [x_t|h] gates GEMM (K = 1280), then pointwise
// LSTM cell. Decoder adds a K=2048 MLP GEMM.
// GEMM mainloops: explicit 2-stage software pipeline with strength-reduced
// addressing (one pointer advance per 2 k-tiles, constant in-loop offsets) so
// loads stay one stage ahead of the WMMAs (wait loadcnt<=10, not 0) without
// per-tile 64-bit address rebuilds.
// All weights (~41 MB bf16) stay resident in the 192 MB L2.
// ---------------------------------------------------------------------------

#define B_  128
#define S_  256
#define D_  256
#define H_  1024
#define T_  128
#define G_  4096   // 4*H

typedef __bf16 bf16_t;
typedef __attribute__((ext_vector_type(16))) __bf16 v16bf;
typedef __attribute__((ext_vector_type(8)))  __bf16 v8bf;
typedef __attribute__((ext_vector_type(8)))  float  v8f;

__device__ __forceinline__ unsigned short f32_to_bf16_rne(float f) {
  union { float f; unsigned int u; } v; v.f = f;
  unsigned int u = v.u;
  u += 0x7FFFu + ((u >> 16) & 1u);          // round-to-nearest-even
  return (unsigned short)(u >> 16);
}

// A-fragment (16x32 bf16, M x K); p already points at
// row base + half*16: 16 contiguous bf16 (k = 16*half + s).
__device__ __forceinline__ v16bf lda(const bf16_t* p) {
  return *(const v16bf*)(p);
}

// B-fragment (32x16 bf16, K x N), W row-major (N x K); p already points at
// W-row base + half*16: 16 contiguous bf16.
__device__ __forceinline__ v16bf ldb(const bf16_t* p) {
  return *(const v16bf*)(p);
}

#define WMMA_BF16(ACC, A, BB) \
  ACC = __builtin_amdgcn_wmma_f32_16x16x32_bf16(false, (A), false, (BB), (short)0, ACC, false, false)

// Load one fragment set (A + 4 B tiles) at element offset OFF from the
// current segment pointers pa / pb0..pb3.
#define LOAD_SET(A_, B0_, B1_, B2_, B3_, OFF) \
  A_  = lda(pa  + (OFF)); \
  B0_ = ldb(pb0 + (OFF)); \
  B1_ = ldb(pb1 + (OFF)); \
  B2_ = ldb(pb2 + (OFF)); \
  B3_ = ldb(pb3 + (OFF));

#define WMMA4(A_, B0_, B1_, B2_, B3_) \
  WMMA_BF16(acc0, A_, B0_); \
  WMMA_BF16(acc1, A_, B1_); \
  WMMA_BF16(acc2, A_, B2_); \
  WMMA_BF16(acc3, A_, B3_);

#define ADV64() { pa += 64; pb0 += 64; pb1 += 64; pb2 += 64; pb3 += 64; }

// ---------------------------------------------------------------------------
// LSTM gates GEMM:  g[B x 4H] = [Ax | Ah] (B x (D+H)) @ [Wx ; Wh]^T
// Block = 256 thr (8 waves). Wave -> 16x64 tile. Block -> 32 x 256.
// grid = (4096/256, 128/32, 2 directions).
// ---------------------------------------------------------------------------
struct LstmSrc {
  const unsigned short* Ax;   // B x D   (row stride ldAx)
  const unsigned short* Ah;   // B x H   (row stride H)
  const unsigned short* Wx;   // 4H x D  bf16
  const unsigned short* Wh;   // 4H x H  bf16
  float*                g;    // B x 4H  f32 (raw, bias added later)
  long                  ldAx;
};

__global__ __launch_bounds__(256) void lstm_gates_gemm(LstmSrc s0, LstmSrc s1) {
  LstmSrc s = (blockIdx.z == 0) ? s0 : s1;
  const bf16_t* Ax = (const bf16_t*)s.Ax;
  const bf16_t* Ah = (const bf16_t*)s.Ah;
  const bf16_t* Wx = (const bf16_t*)s.Wx;
  const bf16_t* Wh = (const bf16_t*)s.Wh;

  const int lane  = threadIdx.x & 31;
  const int wave  = threadIdx.x >> 5;
  const int half  = lane >> 4;
  const int lmod  = lane & 15;
  const int mTile = blockIdx.y * 2 + (wave >> 2);       // 16-row tile index
  const int nBase = blockIdx.x * 256 + (wave & 3) * 64; // 64-col wave tile
  const int mRow  = mTile * 16 + lmod;

  v8f acc0 = {}, acc1 = {}, acc2 = {}, acc3 = {};
  v16bf a0, b00, b01, b02, b03, a1, b10, b11, b12, b13;

  // ---- x part: K = D = 256 -> 8 k-tiles, 2-stage pipeline ----
  const bf16_t* pa  = Ax + (long)mRow * s.ldAx + half * 16;
  const bf16_t* pb0 = Wx + (long)(nBase + lmod) * D_ + half * 16;
  const bf16_t* pb1 = pb0 + 16 * D_;
  const bf16_t* pb2 = pb0 + 32 * D_;
  const bf16_t* pb3 = pb0 + 48 * D_;

  LOAD_SET(a0, b00, b01, b02, b03, 0);                  // tile 0
  #pragma unroll 1
  for (int kt = 0; kt < 6; kt += 2) {
    LOAD_SET(a1, b10, b11, b12, b13, 32);               // tile kt+1
    WMMA4(a0, b00, b01, b02, b03);                      // consume kt
    ADV64();
    LOAD_SET(a0, b00, b01, b02, b03, 0);                // tile kt+2
    WMMA4(a1, b10, b11, b12, b13);                      // consume kt+1
  }
  // x epilogue (tiles 6 in a0, 7) + bridge: preload h tile 0 while the last
  // x-part WMMAs drain.
  LOAD_SET(a1, b10, b11, b12, b13, 32);                 // x tile 7
  pa  = Ah + (long)mRow * H_ + half * 16;               // switch to h part
  pb0 = Wh + (long)(nBase + lmod) * H_ + half * 16;
  pb1 = pb0 + 16 * H_;
  pb2 = pb0 + 32 * H_;
  pb3 = pb0 + 48 * H_;
  WMMA4(a0, b00, b01, b02, b03);                        // x tile 6
  LOAD_SET(a0, b00, b01, b02, b03, 0);                  // h tile 0
  WMMA4(a1, b10, b11, b12, b13);                        // x tile 7

  // ---- h part: K = H = 1024 -> 32 k-tiles ----
  #pragma unroll 1
  for (int kt = 0; kt < 30; kt += 2) {
    LOAD_SET(a1, b10, b11, b12, b13, 32);
    WMMA4(a0, b00, b01, b02, b03);
    ADV64();
    LOAD_SET(a0, b00, b01, b02, b03, 0);
    WMMA4(a1, b10, b11, b12, b13);
  }
  LOAD_SET(a1, b10, b11, b12, b13, 32);                 // h tile 31
  WMMA4(a0, b00, b01, b02, b03);                        // h tile 30
  WMMA4(a1, b10, b11, b12, b13);                        // h tile 31

  // C/D layout: VGPR r, lanes 0-15 -> M = r; lanes 16-31 -> M = 8+r; N = lmod.
  #pragma unroll
  for (int r = 0; r < 8; ++r) {
    const int  row  = mTile * 16 + half * 8 + r;
    const long base = (long)row * G_ + nBase + lmod;
    s.g[base +  0] = acc0[r];
    s.g[base + 16] = acc1[r];
    s.g[base + 32] = acc2[r];
    s.g[base + 48] = acc3[r];
  }
}

// ---------------------------------------------------------------------------
// Pointwise LSTM cell: c' = sig(f)*c + sig(i)*tanh(g); h' = sig(o)*tanh(c')
// grid = (B*H/256, 2 directions). h stored bf16 (next GEMM operand), c f32.
// ---------------------------------------------------------------------------
__global__ __launch_bounds__(256) void lstm_cell_kernel(
    const float* g0, const float* g1,
    const float* bias0, const float* bias1,
    unsigned short* h0, unsigned short* h1,
    float* c0, float* c1) {
  const int dir = blockIdx.y;
  const float* g    = dir ? g1 : g0;
  const float* bias = dir ? bias1 : bias0;
  unsigned short* h = dir ? h1 : h0;
  float* c          = dir ? c1 : c0;

  const int idx = blockIdx.x * 256 + threadIdx.x;  // [0, B*H)
  const int m = idx >> 10;
  const int j = idx & (H_ - 1);
  const float* gr = g + (long)m * G_;

  const float gi = gr[j]           + bias[j];
  const float gf = gr[H_ + j]      + bias[H_ + j];
  const float gg = gr[2 * H_ + j]  + bias[2 * H_ + j];
  const float go = gr[3 * H_ + j]  + bias[3 * H_ + j];

  const float si = 1.0f / (1.0f + __expf(-gi));
  const float sf = 1.0f / (1.0f + __expf(-gf));
  const float so = 1.0f / (1.0f + __expf(-go));

  const float cn = sf * c[idx] + si * tanhf(gg);
  c[idx] = cn;
  h[idx] = f32_to_bf16_rne(so * tanhf(cn));
}

// ---------------------------------------------------------------------------
// Decoder MLP:  out[B x D] = [hf | hb] (B x 2H) @ mlp_W^T + mlp_b
// Writes f32 slice of d_out at timestep t AND bf16 next-step input.
// grid = (B/32); block covers 32 x 256 (full N). Same pipelined mainloop.
// ---------------------------------------------------------------------------
__global__ __launch_bounds__(256) void mlp_decoder_out(
    const unsigned short* hf_, const unsigned short* hb_,
    const unsigned short* Wm_, const float* bias,
    float* out, unsigned short* dec_in, int tstep) {
  const bf16_t* hf = (const bf16_t*)hf_;
  const bf16_t* hb = (const bf16_t*)hb_;
  const bf16_t* Wm = (const bf16_t*)Wm_;   // D x 2H row-major

  const int lane  = threadIdx.x & 31;
  const int wave  = threadIdx.x >> 5;
  const int half  = lane >> 4;
  const int lmod  = lane & 15;
  const int mTile = blockIdx.x * 2 + (wave >> 2);
  const int nBase = (wave & 3) * 64;
  const int mRow  = mTile * 16 + lmod;

  v8f acc0 = {}, acc1 = {}, acc2 = {}, acc3 = {};
  v16bf a0, b00, b01, b02, b03, a1, b10, b11, b12, b13;

  // ---- half 1: A = hf, W cols [0, H) -> 32 k-tiles ----
  const bf16_t* pa  = hf + (long)mRow * H_ + half * 16;
  const bf16_t* pb0 = Wm + (long)(nBase + lmod) * (2 * H_) + half * 16;
  const bf16_t* pb1 = pb0 + 16 * (2 * H_);
  const bf16_t* pb2 = pb0 + 32 * (2 * H_);
  const bf16_t* pb3 = pb0 + 48 * (2 * H_);

  LOAD_SET(a0, b00, b01, b02, b03, 0);
  #pragma unroll 1
  for (int kt = 0; kt < 30; kt += 2) {
    LOAD_SET(a1, b10, b11, b12, b13, 32);
    WMMA4(a0, b00, b01, b02, b03);
    ADV64();
    LOAD_SET(a0, b00, b01, b02, b03, 0);
    WMMA4(a1, b10, b11, b12, b13);
  }
  // epilogue (tiles 30, 31) + bridge to half 2
  LOAD_SET(a1, b10, b11, b12, b13, 32);
  pa  = hb + (long)mRow * H_ + half * 16;               // switch A source
  pb0 = Wm + (long)(nBase + lmod) * (2 * H_) + H_ + half * 16;  // W cols [H,2H)
  pb1 = pb0 + 16 * (2 * H_);
  pb2 = pb0 + 32 * (2 * H_);
  pb3 = pb0 + 48 * (2 * H_);
  WMMA4(a0, b00, b01, b02, b03);
  LOAD_SET(a0, b00, b01, b02, b03, 0);
  WMMA4(a1, b10, b11, b12, b13);

  // ---- half 2: A = hb, W cols [H, 2H) -> 32 k-tiles ----
  #pragma unroll 1
  for (int kt = 0; kt < 30; kt += 2) {
    LOAD_SET(a1, b10, b11, b12, b13, 32);
    WMMA4(a0, b00, b01, b02, b03);
    ADV64();
    LOAD_SET(a0, b00, b01, b02, b03, 0);
    WMMA4(a1, b10, b11, b12, b13);
  }
  LOAD_SET(a1, b10, b11, b12, b13, 32);
  WMMA4(a0, b00, b01, b02, b03);
  WMMA4(a1, b10, b11, b12, b13);

  #pragma unroll
  for (int r = 0; r < 8; ++r) {
    const int row = mTile * 16 + half * 8 + r;
    const int c0 = nBase + lmod, c1 = c0 + 16, c2 = c0 + 32, c3 = c0 + 48;
    const float v0 = acc0[r] + bias[c0];
    const float v1 = acc1[r] + bias[c1];
    const float v2 = acc2[r] + bias[c2];
    const float v3 = acc3[r] + bias[c3];
    float* o = out + ((long)row * T_ + tstep) * D_;
    o[c0] = v0; o[c1] = v1; o[c2] = v2; o[c3] = v3;
    unsigned short* di = dec_in + (long)row * D_;
    di[c0] = f32_to_bf16_rne(v0);
    di[c1] = f32_to_bf16_rne(v1);
    di[c2] = f32_to_bf16_rne(v2);
    di[c3] = f32_to_bf16_rne(v3);
  }
}

// --------------------------- setup kernels ---------------------------------
__global__ void k_f32_to_bf16(const float* __restrict__ src,
                              unsigned short* __restrict__ dst, long n) {
  long i = (long)blockIdx.x * blockDim.x + threadIdx.x;
  const long stride = (long)gridDim.x * blockDim.x;
  for (; i < n; i += stride) dst[i] = f32_to_bf16_rne(src[i]);
}

__global__ void k_zero_u32(unsigned int* p, long n) {
  long i = (long)blockIdx.x * blockDim.x + threadIdx.x;
  const long stride = (long)gridDim.x * blockDim.x;
  for (; i < n; i += stride) p[i] = 0u;
}

__global__ void k_init_decin(const float* __restrict__ x,
                             unsigned short* __restrict__ dec_in) {
  const int i = blockIdx.x * 256 + threadIdx.x;   // B*D = 32768
  const int m = i >> 8, d = i & (D_ - 1);
  dec_in[i] = f32_to_bf16_rne(x[((long)m * S_ + (S_ - 1)) * D_ + d]);
}

// ---------------------------------------------------------------------------
extern "C" void kernel_launch(void* const* d_in, const int* in_sizes, int n_in,
                              void* d_out, int out_size, void* d_ws, size_t ws_size,
                              hipStream_t stream) {
  (void)in_sizes; (void)n_in; (void)out_size; (void)ws_size;

  const float* x         = (const float*)d_in[0];
  // d_in[1] (x_target) is unused by the reference.
  const float* enc_Wih_f = (const float*)d_in[2];
  const float* enc_Whh_f = (const float*)d_in[3];
  const float* enc_b_f   = (const float*)d_in[4];
  const float* enc_Wih_b = (const float*)d_in[5];
  const float* enc_Whh_b = (const float*)d_in[6];
  const float* enc_b_b   = (const float*)d_in[7];
  const float* dec_Wih_f = (const float*)d_in[8];
  const float* dec_Whh_f = (const float*)d_in[9];
  const float* dec_b_f   = (const float*)d_in[10];
  const float* dec_Wih_b = (const float*)d_in[11];
  const float* dec_Whh_b = (const float*)d_in[12];
  const float* dec_b_b   = (const float*)d_in[13];
  const float* mlp_W     = (const float*)d_in[14];
  const float* mlp_b     = (const float*)d_in[15];
  float* out = (float*)d_out;

  // ---- carve workspace (~64 MB) ----
  char* p = (char*)d_ws;
  auto carve = [&](size_t bytes) -> void* {
    void* r = (void*)p; p += (bytes + 255) & ~(size_t)255; return r;
  };
  unsigned short* xbf    = (unsigned short*)carve((size_t)B_ * S_ * D_ * 2);
  unsigned short* wEWihF = (unsigned short*)carve((size_t)G_ * D_ * 2);
  unsigned short* wEWhhF = (unsigned short*)carve((size_t)G_ * H_ * 2);
  unsigned short* wEWihB = (unsigned short*)carve((size_t)G_ * D_ * 2);
  unsigned short* wEWhhB = (unsigned short*)carve((size_t)G_ * H_ * 2);
  unsigned short* wDWihF = (unsigned short*)carve((size_t)G_ * D_ * 2);
  unsigned short* wDWhhF = (unsigned short*)carve((size_t)G_ * H_ * 2);
  unsigned short* wDWihB = (unsigned short*)carve((size_t)G_ * D_ * 2);
  unsigned short* wDWhhB = (unsigned short*)carve((size_t)G_ * H_ * 2);
  unsigned short* wMlp   = (unsigned short*)carve((size_t)D_ * 2 * H_ * 2);
  unsigned short* hF     = (unsigned short*)carve((size_t)B_ * H_ * 2);
  unsigned short* hB     = (unsigned short*)carve((size_t)B_ * H_ * 2);
  float*          cF     = (float*)carve((size_t)B_ * H_ * 4);
  float*          cB     = (float*)carve((size_t)B_ * H_ * 4);
  float*          g0     = (float*)carve((size_t)B_ * G_ * 4);
  float*          g1     = (float*)carve((size_t)B_ * G_ * 4);
  unsigned short* decIn  = (unsigned short*)carve((size_t)B_ * D_ * 2);

  // ---- one-time conversions / state init (per call: deterministic) ----
  k_f32_to_bf16<<<2048, 256, 0, stream>>>(x,         xbf,    (long)B_ * S_ * D_);
  k_f32_to_bf16<<<512,  256, 0, stream>>>(enc_Wih_f, wEWihF, (long)G_ * D_);
  k_f32_to_bf16<<<1024, 256, 0, stream>>>(enc_Whh_f, wEWhhF, (long)G_ * H_);
  k_f32_to_bf16<<<512,  256, 0, stream>>>(enc_Wih_b, wEWihB, (long)G_ * D_);
  k_f32_to_bf16<<<1024, 256, 0, stream>>>(enc_Whh_b, wEWhhB, (long)G_ * H_);
  k_f32_to_bf16<<<512,  256, 0, stream>>>(dec_Wih_f, wDWihF, (long)G_ * D_);
  k_f32_to_bf16<<<1024, 256, 0, stream>>>(dec_Whh_f, wDWhhF, (long)G_ * H_);
  k_f32_to_bf16<<<512,  256, 0, stream>>>(dec_Wih_b, wDWihB, (long)G_ * D_);
  k_f32_to_bf16<<<1024, 256, 0, stream>>>(dec_Whh_b, wDWhhB, (long)G_ * H_);
  k_f32_to_bf16<<<512,  256, 0, stream>>>(mlp_W,     wMlp,   (long)D_ * 2 * H_);
  k_zero_u32<<<128, 256, 0, stream>>>((unsigned int*)hF, (long)B_ * H_ / 2);
  k_zero_u32<<<128, 256, 0, stream>>>((unsigned int*)hB, (long)B_ * H_ / 2);
  k_zero_u32<<<256, 256, 0, stream>>>((unsigned int*)cF, (long)B_ * H_);
  k_zero_u32<<<256, 256, 0, stream>>>((unsigned int*)cB, (long)B_ * H_);
  k_init_decin<<<B_ * D_ / 256, 256, 0, stream>>>(x, decIn);

  const dim3 ggrid(G_ / 256, B_ / 32, 2);   // (16, 4, 2 dirs)
  const dim3 cgrid(B_ * H_ / 256, 2);       // (512, 2 dirs)

  // ---- encoder: forward dir uses t=i, backward dir uses t=S-1-i ----
  for (int i = 0; i < S_; ++i) {
    LstmSrc sf = { xbf + (size_t)i * D_,            hF, wEWihF, wEWhhF, g0, (long)S_ * D_ };
    LstmSrc sb = { xbf + (size_t)(S_ - 1 - i) * D_, hB, wEWihB, wEWhhB, g1, (long)S_ * D_ };
    lstm_gates_gemm<<<ggrid, 256, 0, stream>>>(sf, sb);
    lstm_cell_kernel<<<cgrid, 256, 0, stream>>>(g0, g1, enc_b_f, enc_b_b, hF, hB, cF, cB);
  }

  // ---- decoder: states continue from encoder finals (same buffers) ----
  for (int u = 0; u < T_; ++u) {
    LstmSrc sf = { decIn, hF, wDWihF, wDWhhF, g0, (long)D_ };
    LstmSrc sb = { decIn, hB, wDWihB, wDWhhB, g1, (long)D_ };
    lstm_gates_gemm<<<ggrid, 256, 0, stream>>>(sf, sb);
    lstm_cell_kernel<<<cgrid, 256, 0, stream>>>(g0, g1, dec_b_f, dec_b_b, hF, hB, cF, cB);
    mlp_decoder_out<<<B_ / 32, 256, 0, stream>>>(hF, hB, wMlp, mlp_b, out, decIn, u);
  }
}
